// DependentLatentModel_22746146800387
// MI455X (gfx1250) — matrix operand order repaired and checked by reading, not compiled, not run on hardware
//
#include <hip/hip_runtime.h>
#include <hip/hip_bf16.h>

// ---------------------------------------------------------------------------
// Problem constants (from reference)
// ---------------------------------------------------------------------------
#define Bsz   64
#define Tlen  512
#define Dsz   1536
#define Hsz   30
#define NCOL  128            // 120 gates + a_x + b_x + 6 pad, padded to 128
#define MROWS (Bsz * Tlen)   // 32768

typedef __attribute__((ext_vector_type(16))) __bf16 v16bf;
typedef __attribute__((ext_vector_type(8)))  float  v8f;

// f32 -> bf16 round-to-nearest-even (scalar helper for prep kernel)
__device__ __forceinline__ unsigned short f2bf(float f) {
    unsigned int u = __float_as_uint(f);
    unsigned int r = u + 0x7FFFu + ((u >> 16) & 1u);
    return (unsigned short)(r >> 16);
}

// ---------------------------------------------------------------------------
// Phase 0: pack combined weight matrix, transposed + bf16.
// WcombT[n][d], n in [0,128): n<120 -> Wih[n][d] ; n==120 -> Wa[d] ;
// n==121 -> Wb[d] ; n>=122 -> 0.  (Wih rows have length D+1 = 1537.)
// ---------------------------------------------------------------------------
__global__ __launch_bounds__(256) void prep_wcomb_kernel(
    const float* __restrict__ Wih, const float* __restrict__ Wa,
    const float* __restrict__ Wb, unsigned short* __restrict__ WT) {
    int idx = blockIdx.x * 256 + threadIdx.x;   // over 128*1536
    if (idx >= NCOL * Dsz) return;
    int n = idx / Dsz;
    int d = idx - n * Dsz;
    float v;
    if (n < 120)       v = Wih[n * (Dsz + 1) + d];
    else if (n == 120) v = Wa[d];
    else if (n == 121) v = Wb[d];
    else               v = 0.0f;
    WT[idx] = f2bf(v);
}

// ---------------------------------------------------------------------------
// Phase 1: Xg[32768,128] = X[32768,1536] x Wcomb  via v_wmma_f32_16x16x32_bf16
// Grid: 256 blocks x 256 threads (8 waves). Block owns 128 M-rows; wave w owns
// rows [w*16, w*16+16) x all 128 N, accumulating 8 v8f tiles over 48 K-chunks.
// K-loop manually unrolled x2 with two disjoint A register sets (no rotation
// movs). B fragments use a 3-deep rotating buffer in straight-line code: the
// three buffers are simultaneously live (forced distinct registers) and each
// reload issues 2-3 WMMAs ahead of its consumer, so waits stay partial.
// ---------------------------------------------------------------------------
union BFrag { v16bf v; uint4 q[2]; };

__device__ __forceinline__ BFrag loadB(const unsigned short* __restrict__ brow,
                                       int nt, int kc) {
    BFrag f;
    const uint4* bp = (const uint4*)(brow + (size_t)nt * 16 * Dsz + kc);
    f.q[0] = bp[0];
    f.q[1] = bp[1];
    return f;
}

__device__ __forceinline__ v16bf cvt_afrag(const float4& a0, const float4& a1,
                                           const float4& a2, const float4& a3) {
    union { v16bf v; __bf16 e[16]; } f;
    f.e[0]  = (__bf16)a0.x; f.e[1]  = (__bf16)a0.y;
    f.e[2]  = (__bf16)a0.z; f.e[3]  = (__bf16)a0.w;
    f.e[4]  = (__bf16)a1.x; f.e[5]  = (__bf16)a1.y;
    f.e[6]  = (__bf16)a1.z; f.e[7]  = (__bf16)a1.w;
    f.e[8]  = (__bf16)a2.x; f.e[9]  = (__bf16)a2.y;
    f.e[10] = (__bf16)a2.z; f.e[11] = (__bf16)a2.w;
    f.e[12] = (__bf16)a3.x; f.e[13] = (__bf16)a3.y;
    f.e[14] = (__bf16)a3.z; f.e[15] = (__bf16)a3.w;
    return f.v;
}

#define WMMA_BF16(ACC, AV, BV) \
    __builtin_amdgcn_wmma_f32_16x16x32_bf16(false, (AV), false, (BV), (short)0, (ACC), false, false)

__global__ __launch_bounds__(256) void gemm_wmma_kernel(
    const float* __restrict__ X, const unsigned short* __restrict__ WT,
    float* __restrict__ Xg) {
    const int wave = threadIdx.x >> 5;
    const int lane = threadIdx.x & 31;
    const int half = lane >> 4;      // selects K-half (this code's contiguous order k = 16*half + s)
    const int l15  = lane & 15;

    // A-fragment row for this lane (both lane-halves hold the same M set)
    const int mrow = blockIdx.x * 128 + wave * 16 + l15;
    const float* __restrict__ xrow = X + (size_t)mrow * Dsz;
    // B base for this lane's column slot within each 16-wide N tile
    const unsigned short* __restrict__ brow = WT + (size_t)l15 * Dsz + 16 * half;

    v8f acc[8];
#pragma unroll
    for (int i = 0; i < 8; ++i)
#pragma unroll
        for (int j = 0; j < 8; ++j) acc[i][j] = 0.0f;

    // A layout: lanes 0-15 hold K in {0..15}+kc, lanes 16-31 {16..31}+kc
    const int r0 = 16 * half;        // first run of 8 within the 32-chunk
    const int r1 = 16 * half + 8;    // second run of 8

    // ---- prime pipeline: A chunk for kc = 0 in the "c" register set
    float4 c0 = *(const float4*)(xrow + r0);
    float4 c1 = *(const float4*)(xrow + r0 + 4);
    float4 c2 = *(const float4*)(xrow + r1);
    float4 c3 = *(const float4*)(xrow + r1 + 4);
    float4 n0, n1, n2, n3;           // "next" set, filled inside the loop

    for (int kc = 0; kc < Dsz; kc += 64) {
        // ================= half-iteration 0: consumes c*, prefetches n* ====
        {
            BFrag f0 = loadB(brow, 0, kc);
            BFrag f1 = loadB(brow, 1, kc);
            BFrag f2 = loadB(brow, 2, kc);
            const v16bf av = cvt_afrag(c0, c1, c2, c3);
            {   // prefetch A for kc+32 into the other set (always in range)
                const float* nx = xrow + kc + 32;
                n0 = *(const float4*)(nx + r0);
                n1 = *(const float4*)(nx + r0 + 4);
                n2 = *(const float4*)(nx + r1);
                n3 = *(const float4*)(nx + r1 + 4);
            }
            acc[0] = WMMA_BF16(acc[0], av, f0.v); f0 = loadB(brow, 3, kc);
            acc[1] = WMMA_BF16(acc[1], av, f1.v); f1 = loadB(brow, 4, kc);
            acc[2] = WMMA_BF16(acc[2], av, f2.v); f2 = loadB(brow, 5, kc);
            acc[3] = WMMA_BF16(acc[3], av, f0.v); f0 = loadB(brow, 6, kc);
            acc[4] = WMMA_BF16(acc[4], av, f1.v); f1 = loadB(brow, 7, kc);
            acc[5] = WMMA_BF16(acc[5], av, f2.v);
            acc[6] = WMMA_BF16(acc[6], av, f0.v);
            acc[7] = WMMA_BF16(acc[7], av, f1.v);
        }
        // ================= half-iteration 1: consumes n*, prefetches c* ====
        {
            const int kd = kc + 32;
            BFrag f0 = loadB(brow, 0, kd);
            BFrag f1 = loadB(brow, 1, kd);
            BFrag f2 = loadB(brow, 2, kd);
            const v16bf av = cvt_afrag(n0, n1, n2, n3);
            if (kc + 64 < Dsz) {   // prefetch A for kc+64 back into c*
                const float* nx = xrow + kc + 64;
                c0 = *(const float4*)(nx + r0);
                c1 = *(const float4*)(nx + r0 + 4);
                c2 = *(const float4*)(nx + r1);
                c3 = *(const float4*)(nx + r1 + 4);
            }
            acc[0] = WMMA_BF16(acc[0], av, f0.v); f0 = loadB(brow, 3, kd);
            acc[1] = WMMA_BF16(acc[1], av, f1.v); f1 = loadB(brow, 4, kd);
            acc[2] = WMMA_BF16(acc[2], av, f2.v); f2 = loadB(brow, 5, kd);
            acc[3] = WMMA_BF16(acc[3], av, f0.v); f0 = loadB(brow, 6, kd);
            acc[4] = WMMA_BF16(acc[4], av, f1.v); f1 = loadB(brow, 7, kd);
            acc[5] = WMMA_BF16(acc[5], av, f2.v);
            acc[6] = WMMA_BF16(acc[6], av, f0.v);
            acc[7] = WMMA_BF16(acc[7], av, f1.v);
        }
    }

    // ---- Store D: VGPR r maps to M = r + 8*half, N = nt*16 + l15
    const int mbase = blockIdx.x * 128 + wave * 16 + 8 * half;
#pragma unroll
    for (int nt = 0; nt < 8; ++nt) {
        const int n = nt * 16 + l15;
#pragma unroll
        for (int r = 0; r < 8; ++r)
            Xg[(size_t)(mbase + r) * NCOL + n] = acc[nt][r];
    }
}

// ---------------------------------------------------------------------------
// Phase 2: sequential scan. One wave per batch element. Lane l<30 owns LSTM
// unit l (all 4 gate types), keeps Whh rows in registers, hx broadcast via
// shuffles. Kuma a/b dot products via wave reduction.
// ---------------------------------------------------------------------------
__global__ __launch_bounds__(32) void scan_kernel(
    const float* __restrict__ Xg, const float* __restrict__ u,
    const float* __restrict__ Wih, const float* __restrict__ Whh,
    const float* __restrict__ bih, const float* __restrict__ bhh,
    const float* __restrict__ Wa, const float* __restrict__ ba,
    const float* __restrict__ Wb, const float* __restrict__ bb,
    float* __restrict__ out) {
    const int b    = blockIdx.x;
    const int lane = threadIdx.x;
    const bool act = (lane < Hsz);

    // Per-lane persistent weights
    float whh[4][Hsz];
    float wih_z[4], bias[4];
#pragma unroll
    for (int t4 = 0; t4 < 4; ++t4) {
        const int g = t4 * Hsz + lane;
        wih_z[t4] = act ? Wih[g * (Dsz + 1) + Dsz] : 0.0f;
        bias[t4]  = act ? (bih[g] + bhh[g]) : 0.0f;
#pragma unroll
        for (int k = 0; k < Hsz; ++k)
            whh[t4][k] = act ? Whh[g * Hsz + k] : 0.0f;
    }
    const float wa_h = act ? Wa[Dsz + lane] : 0.0f;
    const float wb_h = act ? Wb[Dsz + lane] : 0.0f;
    const float sba = ba[0], sbb = bb[0];

    float hx = 0.0f, cx = 0.0f;

    for (int t = 0; t < Tlen; ++t) {
        const size_t row = (size_t)b * Tlen + t;

        // Kuma gate dot products with hx, wave-reduced
        float pa = wa_h * hx;
        float pb = wb_h * hx;
#pragma unroll
        for (int off = 16; off > 0; off >>= 1) {
            pa += __shfl_xor(pa, off, 32);
            pb += __shfl_xor(pb, off, 32);
        }
        const float a_pre = Xg[row * NCOL + 120] + pa + sba;
        const float b_pre = Xg[row * NCOL + 121] + pb + sbb;
        // softplus with overflow guard, then clip
        float av = (a_pre > 20.0f) ? a_pre : log1pf(expf(a_pre));
        float bv = (b_pre > 20.0f) ? b_pre : log1pf(expf(b_pre));
        av = fminf(fmaxf(av, 1e-6f), 100.0f);
        bv = fminf(fmaxf(bv, 1e-6f), 100.0f);

        // HardKuma reparameterized sample
        float uu = u[row];
        uu = fminf(fmaxf(uu, 1e-5f), 1.0f - 1e-5f);
        const float s  = powf(1.0f - powf(1.0f - uu, 1.0f / bv), 1.0f / av);
        const float tt = s * 1.2f - 0.1f;               // stretch to [-0.1,1.1]
        const float z  = fminf(fmaxf(tt, 0.0f), 1.0f);  // hard rectify

        if (lane == 0) out[row] = z;

        // LSTM gates: precomputed x-part + z column + bias + Whh @ hx
        float gate[4];
#pragma unroll
        for (int t4 = 0; t4 < 4; ++t4)
            gate[t4] = Xg[row * NCOL + t4 * Hsz + lane] + z * wih_z[t4] + bias[t4];
#pragma unroll
        for (int k = 0; k < Hsz; ++k) {
            const float hk = __shfl(hx, k, 32);
#pragma unroll
            for (int t4 = 0; t4 < 4; ++t4)
                gate[t4] = fmaf(whh[t4][k], hk, gate[t4]);
        }
        const float ig = 1.0f / (1.0f + expf(-gate[0]));
        const float fg = 1.0f / (1.0f + expf(-gate[1]));
        const float gg = tanhf(gate[2]);
        const float og = 1.0f / (1.0f + expf(-gate[3]));
        cx = fg * cx + ig * gg;
        hx = og * tanhf(cx);
        if (!act) { hx = 0.0f; cx = 0.0f; }  // keep idle lanes clean
    }
}

// ---------------------------------------------------------------------------
// Launcher
// ---------------------------------------------------------------------------
extern "C" void kernel_launch(void* const* d_in, const int* in_sizes, int n_in,
                              void* d_out, int out_size, void* d_ws, size_t ws_size,
                              hipStream_t stream) {
    const float* x   = (const float*)d_in[0];
    const float* u   = (const float*)d_in[1];
    const float* Wih = (const float*)d_in[2];
    const float* Whh = (const float*)d_in[3];
    const float* bih = (const float*)d_in[4];
    const float* bhh = (const float*)d_in[5];
    const float* Wa  = (const float*)d_in[6];
    const float* ba  = (const float*)d_in[7];
    const float* Wb  = (const float*)d_in[8];
    const float* bb  = (const float*)d_in[9];
    float* out = (float*)d_out;

    // Workspace layout: [bf16 WcombT 128x1536 | f32 Xg 32768x128]
    unsigned short* WT = (unsigned short*)d_ws;                       // 384 KB
    float* Xg = (float*)((char*)d_ws + (size_t)512 * 1024);           // 16 MB

    // Phase 0: pack weights (128*1536 elems)
    prep_wcomb_kernel<<<(NCOL * Dsz + 255) / 256, 256, 0, stream>>>(
        Wih, Wa, Wb, WT);

    // Phase 1: big bf16 WMMA GEMM over all timesteps
    gemm_wmma_kernel<<<MROWS / 128, 256, 0, stream>>>(x, WT, Xg);

    // Phase 2: latency-bound recurrence, one wave per batch element
    scan_kernel<<<Bsz, 32, 0, stream>>>(Xg, u, Wih, Whh, bih, bhh,
                                        Wa, ba, Wb, bb, out);
}
